// BGATAttention_72267119722740
// MI455X (gfx1250) — compile-verified, matrix operand order checked by reading, not run on hardware
//
#include <hip/hip_runtime.h>
#include <hip/hip_bf16.h>
#include <math.h>

// BGAT attention for MI455X (gfx1250, wave32, WMMA).
// Sizes fixed by the reference: B=8, M=128, N=256, H=8, HD=32, UD=AD=64, ED=16, HID=256.

typedef __attribute__((ext_vector_type(16))) _Float16 v16h;
typedef __attribute__((ext_vector_type(8)))  float    v8f;

__device__ __forceinline__ float leaky(float x) { return x >= 0.f ? x : 0.2f * x; }

// K index held by fragment element e for lane-group g (0: lanes 0-15, 1: lanes 16-31)
// per CDNA5 ISA 16-bit A/B 16x16x32 layout.
__device__ __forceinline__ int kmap(int e, int g) { return (e < 8 ? e : e + 8) + 8 * g; }

// ---------------------------------------------------------------------------
// P(R x 256) = X(R x 64) @ W^T(64 x 256), f16 WMMA, f32 accumulate.
// grid = (R/16)*16 blocks of one wave; W row-major (256 x 64).
// ---------------------------------------------------------------------------
__global__ void __launch_bounds__(32)
bgat_proj_wmma(const float* __restrict__ X, const float* __restrict__ W,
               float* __restrict__ P) {
  const int C = 256, K = 64;
  int bid = blockIdx.x;
  int rt = bid >> 4, ct = bid & 15;
  int lane = threadIdx.x & 31;
  int g = lane >> 4, q = lane & 15;
  int r0 = rt * 16, c0 = ct * 16;
  v8f acc = {};
  for (int kb = 0; kb < K; kb += 32) {
    v16h a, bf;
    int row = r0 + q;
    int col = c0 + q;
#pragma unroll
    for (int e = 0; e < 16; ++e) {
      int k = kb + kmap(e, g);
      a[e]  = (_Float16)X[row * K + k];
      bf[e] = (_Float16)W[col * K + k];
    }
    acc = __builtin_amdgcn_wmma_f32_16x16x32_f16(false, a, false, bf,
                                                 (short)0, acc, false, false);
  }
#pragma unroll
  for (int j = 0; j < 8; ++j)
    P[(r0 + j + 8 * g) * C + c0 + q] = acc[j];
}

// ---------------------------------------------------------------------------
// In-place LayerNorm(32-wide head groups) + LeakyReLU.  One wave per (row, head).
// ---------------------------------------------------------------------------
__global__ void __launch_bounds__(128)
bgat_ln_leaky(float* __restrict__ P, const float* __restrict__ gma,
              const float* __restrict__ bta, int R) {
  int gid = blockIdx.x * (blockDim.x >> 5) + (threadIdx.x >> 5);
  if (gid >= R * 8) return;
  int lane = threadIdx.x & 31;
  int row = gid >> 3, h = gid & 7;
  int c = h * 32 + lane;
  float v = P[row * 256 + c];
  float s = v;
#pragma unroll
  for (int m = 16; m >= 1; m >>= 1) s += __shfl_xor(s, m, 32);
  float mean = s * (1.f / 32.f);
  float d = v - mean;
  float s2 = d * d;
#pragma unroll
  for (int m = 16; m >= 1; m >>= 1) s2 += __shfl_xor(s2, m, 32);
  float var = s2 * (1.f / 32.f);
  float y = d * rsqrtf(var + 1e-5f) * gma[c] + bta[c];
  P[row * 256 + c] = leaky(y);
}

// ---------------------------------------------------------------------------
// Fused per-(b,h,m) kernel: edge projection (WMMA, K=16 zero-padded to 32)
// -> LN -> leaky -> +U +A -> leaky -> dot(av) -> softmax over N
// -> alpha (to ws) and user_heads row.  128 threads = 4 waves.
// Per wave: the two We B-fragments (db=0/1) are loaded once; each edge
// A-fragment then feeds two back-to-back WMMAs.
// ---------------------------------------------------------------------------
__global__ void __launch_bounds__(128)
bgat_score(const float* __restrict__ edge, const float* __restrict__ We,
           const float* __restrict__ ge, const float* __restrict__ be,
           const float* __restrict__ av, const float* __restrict__ U,
           const float* __restrict__ A, float* __restrict__ alpha,
           float* __restrict__ uh) {
  __shared__ float sE[256 * 33];     // padded stride 33 to dodge bank conflicts
  __shared__ float sScore[256];
  __shared__ float sRed[128];
  __shared__ float sU[32];
  __shared__ float sG[32], sBe[32], sAv[32];

  int bid = blockIdx.x;
  int m = bid % 128;
  int h = (bid / 128) % 8;
  int b = bid / (128 * 8);
  int tid = threadIdx.x;
  int w = tid >> 5, lane = tid & 31, g = lane >> 4, q = lane & 15;

  // Phase 0: stage per-head constants into LDS (one wave each).
  if (tid < 32)       sU[tid]        = U[(b * 128 + m) * 256 + h * 32 + tid];
  else if (tid < 64)  sG[tid - 32]   = ge[h * 32 + (tid - 32)];
  else if (tid < 96)  sBe[tid - 64]  = be[h * 32 + (tid - 64)];
  else                sAv[tid - 96]  = av[h * 32 + (tid - 96)];

  const float* Erow = edge + (size_t)(b * 128 + m) * 256 * 16;

  // Per-wave constant B-fragments: We[h, db*16+q, :16], K padded to 32.
  v16h bf0, bf1;
#pragma unroll
  for (int e = 0; e < 16; ++e) {
    int k = kmap(e, g);              // k<16  <=>  e<8 (compile-time foldable)
    bf0[e] = (k < 16) ? (_Float16)We[(h * 32 +      q) * 16 + k] : (_Float16)0.f;
    bf1[e] = (k < 16) ? (_Float16)We[(h * 32 + 16 + q) * 16 + k] : (_Float16)0.f;
  }

  // Phase 1: Eproj[n][d] = edge[b,m,n,:16] . We[h,d,:16] via WMMA.
  for (int nb = w; nb < 16; nb += 4) {
    v16h a;
#pragma unroll
    for (int e = 0; e < 16; ++e) {
      int k = kmap(e, g);
      a[e] = (k < 16) ? (_Float16)Erow[(nb * 16 + q) * 16 + k] : (_Float16)0.f;
    }
    v8f c0 = {}, c1 = {};
    c0 = __builtin_amdgcn_wmma_f32_16x16x32_f16(false, a, false, bf0,
                                                (short)0, c0, false, false);
    c1 = __builtin_amdgcn_wmma_f32_16x16x32_f16(false, a, false, bf1,
                                                (short)0, c1, false, false);
#pragma unroll
    for (int j = 0; j < 8; ++j) {
      int n = nb * 16 + j + 8 * g;
      sE[n * 33 +      q] = c0[j];
      sE[n * 33 + 16 + q] = c1[j];
    }
  }
  __syncthreads();

  // Phase 2: per-n LN + leaky + inter + dot(av) -> score
  for (int n = tid; n < 256; n += 128) {
    float mean = 0.f;
#pragma unroll
    for (int d = 0; d < 32; ++d) mean += sE[n * 33 + d];
    mean *= (1.f / 32.f);
    float var = 0.f;
#pragma unroll
    for (int d = 0; d < 32; ++d) { float df = sE[n * 33 + d] - mean; var += df * df; }
    var *= (1.f / 32.f);
    float inv = rsqrtf(var + 1e-5f);
    const float* Arow = A + (size_t)(b * 256 + n) * 256 + h * 32;
    float acc = 0.f;
#pragma unroll
    for (int d = 0; d < 32; ++d) {
      float ep = (sE[n * 33 + d] - mean) * inv * sG[d] + sBe[d];
      ep = leaky(ep);
      float it = leaky(sU[d] + Arow[d] + ep);
      acc += it * sAv[d];
    }
    sScore[n] = acc * 0.17677669529663687f;  // 1/sqrt(32)
  }
  __syncthreads();

  // Phase 3: softmax over 256
  float v = fmaxf(sScore[tid], sScore[tid + 128]);
  sRed[tid] = v; __syncthreads();
  for (int s = 64; s > 0; s >>= 1) {
    if (tid < s) sRed[tid] = fmaxf(sRed[tid], sRed[tid + s]);
    __syncthreads();
  }
  float mx = sRed[0]; __syncthreads();
  float e0 = expf(sScore[tid] - mx), e1 = expf(sScore[tid + 128] - mx);
  sScore[tid] = e0; sScore[tid + 128] = e1;
  sRed[tid] = e0 + e1; __syncthreads();
  for (int s = 64; s > 0; s >>= 1) {
    if (tid < s) sRed[tid] += sRed[tid + s];
    __syncthreads();
  }
  float inv_total = 1.f / sRed[0]; __syncthreads();
  float* arow = alpha + (size_t)((b * 8 + h) * 128 + m) * 256;
  for (int n = tid; n < 256; n += 128) {
    float al = sScore[n] * inv_total;
    sScore[n] = al;
    arow[n] = al;
  }
  __syncthreads();

  // Phase 4: user_heads[b,h,m,d] = sum_n alpha[n] * A[b,h,n,d]
  {
    int d = tid & 31, chunk = tid >> 5;
    float acc = 0.f;
    for (int n = chunk * 64; n < chunk * 64 + 64; ++n)
      acc += sScore[n] * A[(size_t)(b * 256 + n) * 256 + h * 32 + d];
    sRed[chunk * 32 + d] = acc;
  }
  __syncthreads();
  if (tid < 32) {
    float s = sRed[tid] + sRed[32 + tid] + sRed[64 + tid] + sRed[96 + tid];
    uh[(size_t)(b * 128 + m) * 256 + h * 32 + tid] = s;   // cat layout
  }
}

// ---------------------------------------------------------------------------
// ant_heads(N x 32) = alpha^T(N x 128) @ U(128 x 32) per (b,h); WMMA K=128.
// One wave per 16x16 output tile; output stored in cat layout (b*N+n, h*32+d).
// ---------------------------------------------------------------------------
__global__ void __launch_bounds__(32)
bgat_ant_heads(const float* __restrict__ alpha, const float* __restrict__ U,
               float* __restrict__ ah) {
  int bid = blockIdx.x;
  int tile = bid & 31;
  int h = (bid >> 5) & 7;
  int b = bid >> 8;
  int nb = tile & 15, db = tile >> 4;
  int lane = threadIdx.x & 31, g = lane >> 4, q = lane & 15;
  const float* al = alpha + (size_t)((b * 8 + h) * 128) * 256;
  const float* Ub = U + (size_t)(b * 128) * 256 + h * 32;
  v8f acc = {};
  for (int kb = 0; kb < 4; ++kb) {
    v16h a, bf;
#pragma unroll
    for (int e = 0; e < 16; ++e) {
      int mi = kb * 32 + kmap(e, g);                 // K index = user row m
      a[e]  = (_Float16)al[mi * 256 + nb * 16 + q];  // A-frag row n, col m (alpha^T)
      bf[e] = (_Float16)Ub[(size_t)mi * 256 + db * 16 + q];
    }
    acc = __builtin_amdgcn_wmma_f32_16x16x32_f16(false, a, false, bf,
                                                 (short)0, acc, false, false);
  }
#pragma unroll
  for (int j = 0; j < 8; ++j) {
    int n = nb * 16 + j + 8 * g;
    ah[(size_t)(b * 256 + n) * 256 + h * 32 + db * 16 + q] = acc[j];
  }
}

// ---------------------------------------------------------------------------
// out = relu(heads_cat + leaky(X @ Wr^T + br)); one thread per output element.
// ---------------------------------------------------------------------------
__global__ void __launch_bounds__(256)
bgat_out_mlp(const float* __restrict__ X, const float* __restrict__ Wr,
             const float* __restrict__ br, const float* __restrict__ heads,
             float* __restrict__ out, int total) {
  int idx = blockIdx.x * blockDim.x + threadIdx.x;
  if (idx >= total) return;
  int r = idx >> 8, c = idx & 255;
  const float* xr = X + r * 64;
  const float* wr = Wr + c * 64;
  float acc = br[c];
#pragma unroll 8
  for (int k = 0; k < 64; ++k) acc += xr[k] * wr[k];
  out[idx] = fmaxf(heads[idx] + leaky(acc), 0.f);
}

// ---------------------------------------------------------------------------
extern "C" void kernel_launch(void* const* d_in, const int* in_sizes, int n_in,
                              void* d_out, int out_size, void* d_ws, size_t ws_size,
                              hipStream_t stream) {
  (void)in_sizes; (void)n_in; (void)out_size; (void)ws_size;
  const float* user = (const float*)d_in[0];
  const float* ant  = (const float*)d_in[1];
  const float* edge = (const float*)d_in[2];
  const float* Wu   = (const float*)d_in[3];
  const float* gu   = (const float*)d_in[4];
  const float* bu   = (const float*)d_in[5];
  const float* Wa   = (const float*)d_in[6];
  const float* ga   = (const float*)d_in[7];
  const float* ba   = (const float*)d_in[8];
  const float* We   = (const float*)d_in[9];
  const float* ge   = (const float*)d_in[10];
  const float* be   = (const float*)d_in[11];
  const float* av   = (const float*)d_in[12];
  const float* Wru  = (const float*)d_in[13];
  const float* bru  = (const float*)d_in[14];
  const float* Wra  = (const float*)d_in[15];
  const float* bra  = (const float*)d_in[16];

  float* ws    = (float*)d_ws;
  float* U     = ws;                        //  1024*256 (proj -> LN in place)
  float* A     = U + 1024 * 256;            //  2048*256
  float* alpha = A + 2048 * 256;            //  8*8*128*256
  float* uh    = alpha + 8 * 8 * 128 * 256; //  1024*256 (cat layout)
  float* ah    = uh + 1024 * 256;           //  2048*256 (cat layout)
  float* uout  = (float*)d_out;             //  1024*256
  float* aout  = uout + 1024 * 256;         //  2048*256

  bgat_proj_wmma<<<1024, 32, 0, stream>>>(user, Wu, U);   // (1024/16)*(256/16) tiles
  bgat_proj_wmma<<<2048, 32, 0, stream>>>(ant,  Wa, A);   // (2048/16)*(256/16) tiles
  bgat_ln_leaky<<<2048, 128, 0, stream>>>(U, gu, bu, 1024);
  bgat_ln_leaky<<<4096, 128, 0, stream>>>(A, ga, ba, 2048);
  bgat_score<<<8 * 8 * 128, 128, 0, stream>>>(edge, We, ge, be, av, U, A, alpha, uh);
  bgat_ant_heads<<<8 * 8 * 32, 32, 0, stream>>>(alpha, U, ah);
  bgat_out_mlp<<<1024, 256, 0, stream>>>(user, Wru, bru, uh, uout, 1024 * 256);
  bgat_out_mlp<<<2048, 256, 0, stream>>>(ant,  Wra, bra, ah, aout, 2048 * 256);
}
